// CoordAtt_84988812853764
// MI455X (gfx1250) — matrix operand order repaired and checked by reading, not compiled, not run on hardware
//
#include <hip/hip_runtime.h>

typedef __attribute__((ext_vector_type(16))) _Float16 v16h;
typedef __attribute__((ext_vector_type(8)))  float    v8f;

#define NB   16
#define CCH  256
#define HH   64
#define WW   64
#define MIP  8
#define HWSZ (HH * WW)      // 4096 per plane
#define POS  (HH + WW)      // 128 pooled positions per (n,c)

// ---------------------------------------------------------------------------
// Stage A: xs = x + residual (on the fly); row means (x_h) and column means
// (x_w) of each 64x64 plane -> y0[n][c][128] = concat(x_h, x_w).
// One block per (n,c) plane. LDS tile padded to stride 65 -> conflict-free
// row and column reductions.
// ---------------------------------------------------------------------------
__global__ __launch_bounds__(256) void coordatt_pool(const float* __restrict__ x,
                                                     const float* __restrict__ res,
                                                     float* __restrict__ y0) {
    __shared__ float tile[64 * 65];
    const int plane  = blockIdx.x;                 // n*CCH + c
    const size_t base = (size_t)plane * HWSZ;
    const int t = threadIdx.x;

#pragma unroll
    for (int i = 0; i < 4; ++i) {
        const int e   = i * 1024 + t * 4;          // element offset in plane
        const int row = e >> 6;
        const int col = e & 63;
        const float4 xv = *(const float4*)(x   + base + e);
        const float4 rv = *(const float4*)(res + base + e);
        float* dst = &tile[row * 65 + col];
        dst[0] = xv.x + rv.x;
        dst[1] = xv.y + rv.y;
        dst[2] = xv.z + rv.z;
        dst[3] = xv.w + rv.w;
    }
    __syncthreads();

    if (t < 64) {                                  // mean over w -> x_h[row]
        float s = 0.f;
#pragma unroll 8
        for (int c0 = 0; c0 < 64; ++c0) s += tile[t * 65 + c0];
        y0[(size_t)plane * POS + t] = s * (1.f / 64.f);
    } else if (t < 128) {                          // mean over h -> x_w[col]
        const int c0 = t - 64;
        float s = 0.f;
#pragma unroll 8
        for (int r = 0; r < 64; ++r) s += tile[r * 65 + c0];
        y0[(size_t)plane * POS + 64 + c0] = s * (1.f / 64.f);
    }
}

// ---------------------------------------------------------------------------
// Stage B: channel squeeze y1[n][m][pos] = hswish(BN(y0[n][:][pos] . w1[m][:]
// + b1[m])).  Per n: [128 pos x 256 c] x [256 c x 8 m] GEMM done with
// v_wmma_f32_16x16x32_f16, one wave per 16-pos tile, K fully unrolled over
// c=256 (8 WMMA ops).  N-dim (m) padded 8 -> 16: B columns l>=8 are loaded
// from a clamped address (l&7) and zeroed with a branchless lane select so
// EXEC stays all-1s around the WMMAs (no saveexec divergence).
// A layout (16x32 f16): lane g=lane>>4, row = lane&15; half hi -> K =
//   g*8 + hi + (hi>=8 ? 8 : 0).
// B layout (32x16 f16): col = lane&15; half hi -> K = g*8 + hi + (hi>=8 ? 8 : 0).
// D layout (16x16 f32): lane -> N = lane&15, VGPR v -> M = (lane>>4)*8 + v.
// ---------------------------------------------------------------------------
__global__ __launch_bounds__(32) void coordatt_squeeze_wmma(
    const float* __restrict__ y0, const float* __restrict__ w1,
    const float* __restrict__ b1, const float* __restrict__ bn_gamma,
    const float* __restrict__ bn_beta, const float* __restrict__ bn_mean,
    const float* __restrict__ bn_var, float* __restrict__ y1) {
    const int n     = blockIdx.x >> 3;
    const int tilei = blockIdx.x & 7;              // 8 pos-tiles of 16
    const int lane  = threadIdx.x;
    const int g     = lane >> 4;
    const int l     = lane & 15;
    const int posA  = tilei * 16 + l;              // A-matrix row (pos)
    const float* aBase  = y0 + (size_t)n * CCH * POS + posA;
    const float* w1row  = w1 + (l & 7) * CCH;      // clamped: always in-bounds
    const bool   mvalid = (l < MIP);

    v8f acc = {};
#pragma unroll
    for (int k0 = 0; k0 < CCH; k0 += 32) {
        v16h a, b;
#pragma unroll
        for (int hi = 0; hi < 16; ++hi) {
            const int kA = g * 8 + hi + (hi >= 8 ? 8 : 0);
            a[hi] = (_Float16)aBase[(k0 + kA) * POS];
            const int kB = g * 8 + hi + (hi >= 8 ? 8 : 0);
            const float wv = w1row[k0 + kB];       // unconditional load
            b[hi] = (_Float16)(mvalid ? wv : 0.f); // branchless v_cndmask
        }
        acc = __builtin_amdgcn_wmma_f32_16x16x32_f16(
            /*neg_a=*/false, a, /*neg_b=*/false, b,
            /*c_mod=*/(short)0, acc, /*reuse_a=*/false, /*reuse_b=*/false);
    }

    if (mvalid) {                                  // valid m columns only
        const int m   = l;
        const float inv = bn_gamma[m] * rsqrtf(bn_var[m] + 1e-5f);
        const float bia = b1[m], mu = bn_mean[m], bet = bn_beta[m];
#pragma unroll
        for (int v = 0; v < 8; ++v) {
            const int pos = tilei * 16 + g * 8 + v;
            float val = acc[v] + bia;
            val = (val - mu) * inv + bet;
            float t6 = fminf(fmaxf(val + 3.f, 0.f), 6.f);   // h-swish
            val = val * t6 * (1.f / 6.f);
            y1[((size_t)n * MIP + m) * POS + pos] = val;
        }
    }
}

// ---------------------------------------------------------------------------
// Stage C: excite. a = sigmoid(sum_m y1[n][m][pos] * wsel[o][m] + bsel[o]).
// pos < 64 uses (w2,b2) -> a_h[n][o][h]; pos >= 64 uses (w3,b3) -> a_w[n][o][w].
// One 128-thread block per (n,o); coalesced along pos.
// ---------------------------------------------------------------------------
__global__ __launch_bounds__(128) void coordatt_excite(
    const float* __restrict__ y1, const float* __restrict__ w2,
    const float* __restrict__ b2, const float* __restrict__ w3,
    const float* __restrict__ b3, float* __restrict__ ah,
    float* __restrict__ aw) {
    const int blk = blockIdx.x;                    // o + CCH*n
    const int n   = blk >> 8;
    const int o   = blk & 255;
    const int pos = threadIdx.x;                   // 0..127
    const float* y1n  = y1 + (size_t)n * MIP * POS;
    const float* wsel = (pos < HH) ? w2 : w3;
    float s = (pos < HH) ? b2[o] : b3[o];
#pragma unroll
    for (int m = 0; m < MIP; ++m)
        s += y1n[m * POS + pos] * wsel[o * MIP + m];
    const float a = 1.f / (1.f + __expf(-s));
    if (pos < HH) ah[((size_t)n * CCH + o) * HH + pos]        = a;
    else          aw[((size_t)n * CCH + o) * WW + (pos - HH)] = a;
}

// ---------------------------------------------------------------------------
// Stage D: out = 2*xs*wei + 2*res*(1-wei), wei = a_h[n,c,h] * a_w[n,c,w].
// Pure streaming pass, float4 per thread; x/res re-reads mostly hit L2
// (134 MB < 192 MB).  global_prefetch_b8 pulls the next tile while the
// current float4 loads are in flight.
// ---------------------------------------------------------------------------
__global__ __launch_bounds__(256) void coordatt_apply(
    const float* __restrict__ x, const float* __restrict__ res,
    const float* __restrict__ ah, const float* __restrict__ aw,
    float* __restrict__ out) {
    const size_t e4  = (size_t)blockIdx.x * 256 + threadIdx.x;
    const size_t idx = e4 * 4;
    const int    wc  = (int)(idx & 63);
    const int    hr  = (int)((idx >> 6) & 63);
    const size_t nc  = idx >> 12;                  // n*CCH + c

    // speculative prefetch of data one block-stride ahead (safe: dropped on
    // translation failure)
    __builtin_prefetch(x   + idx + 256 * 4, 0, 0);
    __builtin_prefetch(res + idx + 256 * 4, 0, 0);

    const float  a_h = ah[nc * HH + hr];
    const float4 a_w = *(const float4*)(aw + nc * WW + wc);
    const float4 xv  = *(const float4*)(x   + idx);
    const float4 rv  = *(const float4*)(res + idx);

    float4 o;
    {
        float wei = a_h * a_w.x, xs = xv.x + rv.x;
        o.x = 2.f * (xs * wei + rv.x * (1.f - wei));
    }
    {
        float wei = a_h * a_w.y, xs = xv.y + rv.y;
        o.y = 2.f * (xs * wei + rv.y * (1.f - wei));
    }
    {
        float wei = a_h * a_w.z, xs = xv.z + rv.z;
        o.z = 2.f * (xs * wei + rv.z * (1.f - wei));
    }
    {
        float wei = a_h * a_w.w, xs = xv.w + rv.w;
        o.w = 2.f * (xs * wei + rv.w * (1.f - wei));
    }
    *(float4*)(out + idx) = o;
}

// ---------------------------------------------------------------------------
extern "C" void kernel_launch(void* const* d_in, const int* in_sizes, int n_in,
                              void* d_out, int out_size, void* d_ws, size_t ws_size,
                              hipStream_t stream) {
    const float* x        = (const float*)d_in[0];
    const float* residual = (const float*)d_in[1];
    const float* w1       = (const float*)d_in[2];
    const float* b1       = (const float*)d_in[3];
    const float* bn_gamma = (const float*)d_in[4];
    const float* bn_beta  = (const float*)d_in[5];
    const float* bn_mean  = (const float*)d_in[6];
    const float* bn_var   = (const float*)d_in[7];
    const float* w2       = (const float*)d_in[8];
    const float* b2       = (const float*)d_in[9];
    const float* w3       = (const float*)d_in[10];
    const float* b3       = (const float*)d_in[11];
    float* out = (float*)d_out;

    // Workspace partition (sizes in bytes):
    //   y0: 16*256*128*4 = 2,097,152
    //   y1: 16*8*128*4   =    65,536
    //   ah: 16*256*64*4  = 1,048,576 (4,194,304 reserved)
    //   aw: 16*256*64*4  = 1,048,576   (ws span ~7.4 MB)
    char* ws = (char*)d_ws;
    float* y0 = (float*)(ws);
    float* y1 = (float*)(ws + 2097152);
    float* ah = (float*)(ws + 2097152 + 65536);
    float* aw = (float*)(ws + 2097152 + 65536 + 4194304);

    // Stage A: pooled means (reads 134 MB from HBM, warms L2 for stage D)
    coordatt_pool<<<NB * CCH, 256, 0, stream>>>(x, residual, y0);

    // Stage B: WMMA channel squeeze + BN + h-swish (128 waves, 8 WMMA each)
    coordatt_squeeze_wmma<<<NB * 8, 32, 0, stream>>>(
        y0, w1, b1, bn_gamma, bn_beta, bn_mean, bn_var, y1);

    // Stage C: excite + sigmoid
    coordatt_excite<<<NB * CCH, 128, 0, stream>>>(y1, w2, b2, w3, b3, ah, aw);

    // Stage D: final streaming elementwise (writes all of d_out)
    coordatt_apply<<<(NB * CCH * HWSZ / 4) / 256, 256, 0, stream>>>(
        x, residual, ah, aw, out);
}
